// DLRM_36292473651204
// MI455X (gfx1250) — matrix operand order repaired and checked by reading, not compiled, not run on hardware
//
#include <hip/hip_runtime.h>
#include <hip/hip_bf16.h>

typedef __attribute__((ext_vector_type(16))) _Float16 v16h;
typedef __attribute__((ext_vector_type(8)))  _Float16 v8h;
typedef __attribute__((ext_vector_type(8)))  float    v8f;

#define B_ 4096
#define F_ 26
#define L_ 10
#define V_ 100000
#define D_ 128

#if defined(__gfx1250__) && __has_builtin(__builtin_amdgcn_tensor_load_to_lds)
#define HAVE_TDM 1
#else
#define HAVE_TDM 0
#endif

static __device__ inline v16h join8(v8h lo, v8h hi) {
  union { v16h v; v8h h[2]; } u;
  u.h[0] = lo; u.h[1] = hi;
  return u.v;
}

#if HAVE_TDM
typedef unsigned int v4u_ __attribute__((ext_vector_type(4)));
typedef int          v8i_ __attribute__((ext_vector_type(8)));
typedef int          v4i_ __attribute__((ext_vector_type(4)));
typedef __attribute__((address_space(3))) _Float16 lds_f16_t;

// TDM: DMA a 128x32-half 2D tile (row stride = Ka halves) from global into LDS at
// byte offset lds_off, hardware-padded to a 40-half (80B) LDS row pitch:
// pad_interval=3 (16 dwords = one 64B tile row), pad_amount=3 (4 dwords = 16B).
static __device__ inline void tdm_load_tile(const _Float16* gsrc, unsigned lds_off,
                                            int stride_elems) {
  unsigned long long ga = (unsigned long long)gsrc;
  v4u_ g0;
  g0[0] = 1u;                                                  // count=1, user, no gather
  g0[1] = lds_off;                                             // lds_addr [63:32]
  g0[2] = (unsigned)ga;                                        // global_addr lo
  g0[3] = ((unsigned)(ga >> 32) & 0x1FFFFFFu) | (2u << 30);    // addr[56:32], type=2
  unsigned td0 = 1u << 30, td1 = 1u << 30;                     // huge tensor dims: no OOB
  v8i_ g1;
  g1[0] = (int)((1u << 16)      // data_size = 2 bytes
              | (1u << 20)      // pad_enable
              | (3u << 22)      // pad_interval: 16 dwords
              | (3u << 25));    // pad_amount: 4 dwords
  g1[1] = (int)((td0 & 0xFFFFu) << 16);                        // atomic_addr=0 | td0.lo
  g1[2] = (int)((td0 >> 16) | ((td1 & 0xFFFFu) << 16));        // td0.hi | td1.lo
  g1[3] = (int)((td1 >> 16) | (32u << 16));                    // td1.hi | tile_dim0=32
  g1[4] = 128;                                                 // tile_dim1=128, tile_dim2=0
  g1[5] = stride_elems;                                        // tensor_dim0_stride lo
  g1[6] = 0;                                                   // stride hi | dim1_stride lo
  g1[7] = 0;
  v4i_ zz = {0, 0, 0, 0};
  v8i_ z8 = {0, 0, 0, 0, 0, 0, 0, 0};
  // 6-arg form on this toolchain: (g0, g1, g2, g3, extra, cpol); 2D tile -> rest zero
  __builtin_amdgcn_tensor_load_to_lds(g0, g1, zz, zz, z8, 0);
}
#endif

// ---------- weight convert: W (K x N) f32 -> Wt (N x Kp) f16, zero-pad K..Kp ----------
__global__ void convert_wt_kernel(const float* __restrict__ W, _Float16* __restrict__ Wt,
                                  int K, int Kp, int N) {
  int tid = blockIdx.x * blockDim.x + threadIdx.x;
  int n = tid / Kp;
  int k = tid % Kp;
  float v = (k < K) ? W[(size_t)k * N + n] : 0.0f;
  Wt[(size_t)n * Kp + k] = (_Float16)v;
}

// ---------- dense input 4096x13 f32 -> x0 4096x32 f16 (zero-padded) ----------
__global__ void convert_input_kernel(const float* __restrict__ x, _Float16* __restrict__ x0) {
  int tid = blockIdx.x * blockDim.x + threadIdx.x;
  int b = tid >> 5;
  int k = tid & 31;
  float v = (k < 13) ? x[b * 13 + k] : 0.0f;
  x0[tid] = (_Float16)v;
}

// ---------- embedding gather + L-pool: z[b][1+f][:] = sum_l tables[f, idx, :] ----------
__global__ void embed_kernel(const float* __restrict__ tables, const int* __restrict__ indices,
                             _Float16* __restrict__ z) {
  __shared__ int sidx[L_];
  int f = blockIdx.x;
  int b = blockIdx.y;
  int d = threadIdx.x;
  if (d < L_) sidx[d] = indices[((size_t)b * F_ + f) * L_ + d];
  __syncthreads();
  float s = 0.f;
#pragma unroll
  for (int l = 0; l < L_; ++l) {
    size_t row = (size_t)f * V_ + (size_t)sidx[l];
    s += tables[row * D_ + d];
  }
  z[((size_t)b * 27 + 1 + f) * D_ + d] = (_Float16)s;
}

__global__ void copy_dense_to_z(const _Float16* __restrict__ a3, _Float16* __restrict__ z) {
  int b = blockIdx.x, d = threadIdx.x;
  z[((size_t)b * 27) * D_ + d] = a3[(size_t)b * D_ + d];
}

// ---------- generic f16 WMMA GEMM: C = act(A(MxKa) * Bt(NxKa)^T + bias) ----------
__global__ __launch_bounds__(256) void gemm_kernel(const _Float16* __restrict__ A,
                                                   const _Float16* __restrict__ Bt,
                                                   const float* __restrict__ bias,
                                                   _Float16* __restrict__ C,
                                                   int N, int Ka, int relu) {
  int n0 = blockIdx.x * 128;
  int m0 = blockIdx.y * 128;
  int t = threadIdx.x;
  int w = t >> 5, lane = t & 31;
  int wm = (w >> 1) * 32;  // 4 wave rows x 32
  int wn = (w & 1) * 64;   // 2 wave cols x 64
  int lr = lane & 15;
  int hi16 = (lane < 16) ? 0 : 1;

  v8f zero = {};
  v8f acc[2][4];
#pragma unroll
  for (int i = 0; i < 2; ++i)
#pragma unroll
    for (int j = 0; j < 4; ++j) acc[i][j] = zero;

#if HAVE_TDM
  // double-buffered LDS tiles, filled by the Tensor Data Mover
  __shared__ _Float16 As[2][128][40];
  __shared__ _Float16 Bs[2][128][40];
  unsigned baseA = (unsigned)(unsigned long long)(lds_f16_t*)&As[0][0][0];
  unsigned baseB = (unsigned)(unsigned long long)(lds_f16_t*)&Bs[0][0][0];
  const unsigned bufBytes = 128u * 40u * 2u;
  int iters = Ka >> 5;
  if (t < 32) {  // wave 0 drives the TDM
    tdm_load_tile(A + (size_t)m0 * Ka, baseA, Ka);
    tdm_load_tile(Bt + (size_t)n0 * Ka, baseB, Ka);
  }
  for (int it = 0; it < iters; ++it) {
    int cur = it & 1;
    if (t < 32) __builtin_amdgcn_s_wait_tensorcnt(0);  // tile `it` landed in LDS
    __syncthreads();                                    // ...and visible to all waves
    if (t < 32 && it + 1 < iters) {                     // DMA tile it+1 during compute
      int k0n = (it + 1) << 5;
      tdm_load_tile(A + (size_t)m0 * Ka + k0n, baseA + (unsigned)(cur ^ 1) * bufBytes, Ka);
      tdm_load_tile(Bt + (size_t)n0 * Ka + k0n, baseB + (unsigned)(cur ^ 1) * bufBytes, Ka);
    }
    v16h af[2], bf[4];
    int koff = hi16 * 8;
#pragma unroll
    for (int ms = 0; ms < 2; ++ms) {
      int row = wm + ms * 16 + lr;
      af[ms] = join8(*(const v8h*)&As[cur][row][koff], *(const v8h*)&As[cur][row][koff + 16]);
    }
    int krow = hi16 * 8;
#pragma unroll
    for (int ns = 0; ns < 4; ++ns) {
      int col = wn + ns * 16 + lr;
      bf[ns] = join8(*(const v8h*)&Bs[cur][col][krow], *(const v8h*)&Bs[cur][col][krow + 16]);
    }
#pragma unroll
    for (int ms = 0; ms < 2; ++ms)
#pragma unroll
      for (int ns = 0; ns < 4; ++ns)
        acc[ms][ns] = __builtin_amdgcn_wmma_f32_16x16x32_f16(
            false, af[ms], false, bf[ns], (short)0, acc[ms][ns], false, false);
    __syncthreads();  // all waves done reading buf `cur` before TDM refills it next round
  }
#else
  // fallback: synchronous global->LDS staging
  __shared__ _Float16 As[128][40];
  __shared__ _Float16 Bs[128][40];
  for (int k0 = 0; k0 < Ka; k0 += 32) {
    __syncthreads();
#pragma unroll
    for (int it = 0; it < 2; ++it) {
      int e = (t + it * 256) * 8;
      int r = e >> 5, c = e & 31;
      *(uint4*)&As[r][c] = *(const uint4*)(A + (size_t)(m0 + r) * Ka + k0 + c);
      *(uint4*)&Bs[r][c] = *(const uint4*)(Bt + (size_t)(n0 + r) * Ka + k0 + c);
    }
    __syncthreads();
    v16h af[2], bf[4];
    int koff = hi16 * 8;
#pragma unroll
    for (int ms = 0; ms < 2; ++ms) {
      int row = wm + ms * 16 + lr;
      af[ms] = join8(*(const v8h*)&As[row][koff], *(const v8h*)&As[row][koff + 16]);
    }
    int krow = hi16 * 8;
#pragma unroll
    for (int ns = 0; ns < 4; ++ns) {
      int col = wn + ns * 16 + lr;
      bf[ns] = join8(*(const v8h*)&Bs[col][krow], *(const v8h*)&Bs[col][krow + 16]);
    }
#pragma unroll
    for (int ms = 0; ms < 2; ++ms)
#pragma unroll
      for (int ns = 0; ns < 4; ++ns)
        acc[ms][ns] = __builtin_amdgcn_wmma_f32_16x16x32_f16(
            false, af[ms], false, bf[ns], (short)0, acc[ms][ns], false, false);
  }
#endif

  // epilogue: bias + ReLU, store f16 per C/D layout (lane = col, VGPR r = row)
#pragma unroll
  for (int ms = 0; ms < 2; ++ms) {
#pragma unroll
    for (int ns = 0; ns < 4; ++ns) {
      int col = n0 + wn + ns * 16 + lr;
      int rbase = m0 + wm + ms * 16 + hi16 * 8;
      float bv = bias[col];
#pragma unroll
      for (int r = 0; r < 8; ++r) {
        float v = acc[ms][ns][r] + bv;
        if (relu) v = fmaxf(v, 0.f);
        C[(size_t)(rbase + r) * N + col] = (_Float16)v;
      }
    }
  }
}

// ---------- interaction: per-sample z z^T upper triangle, fused concat into feats ----------
static __device__ inline v16h loadA_z(const _Float16* zb, int mbase, int kb, int lane) {
  int row = mbase + (lane & 15);
  int koff = (lane < 16) ? 0 : 8;
  v16h a = {};
  if (row < 27) {
    const _Float16* p = zb + (size_t)row * 128 + kb + koff;
    a = join8(*(const v8h*)p, *(const v8h*)(p + 16));
  }
  return a;
}
static __device__ inline v16h loadB_z(const _Float16* zb, int nbase, int kb, int lane) {
  int col = nbase + (lane & 15);
  int krow = kb + ((lane < 16) ? 0 : 8);
  v16h b = {};
  if (col < 27) {
    const _Float16* p = zb + (size_t)col * 128 + krow;
    b = join8(*(const v8h*)p, *(const v8h*)(p + 16));
  }
  return b;
}
static __device__ inline void store_tri(v8f acc, int mbase, int nbase, int lane,
                                        _Float16* __restrict__ fb) {
  int j = nbase + (lane & 15);
  int ib = mbase + ((lane < 16) ? 0 : 8);
#pragma unroll
  for (int r = 0; r < 8; ++r) {
    int i = ib + r;
    if (i < j && j < 27) {
      int idx = i * 26 - (i * (i - 1)) / 2 + (j - i - 1);  // row-major triu(k=1)
      fb[128 + idx] = (_Float16)acc[r];
    }
  }
}

__global__ __launch_bounds__(256) void interact_kernel(const _Float16* __restrict__ z,
                                                       const _Float16* __restrict__ a3,
                                                       _Float16* __restrict__ feats) {
  int w = threadIdx.x >> 5, lane = threadIdx.x & 31;
  int b = blockIdx.x * 8 + w;  // one wave per sample
  const _Float16* zb = z + (size_t)b * 27 * 128;
  _Float16* fb = feats + (size_t)b * 480;

  // feats[0:128] = embedded_dense; feats[479] = 0 pad
  *(uint2*)(fb + lane * 4) = *(const uint2*)(a3 + (size_t)b * 128 + lane * 4);
  if (lane == 0) fb[479] = (_Float16)0.f;

  v8f a00 = {}, a01 = {}, a11 = {};
#pragma unroll
  for (int kb = 0; kb < 128; kb += 32) {
    v16h A0 = loadA_z(zb, 0, kb, lane);
    v16h A1 = loadA_z(zb, 16, kb, lane);
    v16h B0 = loadB_z(zb, 0, kb, lane);
    v16h B1 = loadB_z(zb, 16, kb, lane);
    a00 = __builtin_amdgcn_wmma_f32_16x16x32_f16(false, A0, false, B0, (short)0, a00, false, false);
    a01 = __builtin_amdgcn_wmma_f32_16x16x32_f16(false, A0, false, B1, (short)0, a01, false, false);
    a11 = __builtin_amdgcn_wmma_f32_16x16x32_f16(false, A1, false, B1, (short)0, a11, false, false);
  }
  store_tri(a00, 0, 0, lane, fb);
  store_tri(a01, 0, 16, lane, fb);
  store_tri(a11, 16, 16, lane, fb);
}

// ---------- final 256 -> 1 layer: wave-reduction dot product ----------
__global__ void final_kernel(const _Float16* __restrict__ A, const float* __restrict__ W,
                             const float* __restrict__ bias, float* __restrict__ out) {
  int row = blockIdx.x * 8 + (threadIdx.x >> 5);
  int lane = threadIdx.x & 31;
  const _Float16* ar = A + (size_t)row * 256;
  float s = 0.f;
#pragma unroll
  for (int k = 0; k < 8; ++k) s += (float)ar[lane + k * 32] * W[lane + k * 32];
#pragma unroll
  for (int o = 16; o > 0; o >>= 1) s += __shfl_down(s, o, 32);
  if (lane == 0) out[row] = s + bias[0];
}

extern "C" void kernel_launch(void* const* d_in, const int* in_sizes, int n_in,
                              void* d_out, int out_size, void* d_ws, size_t ws_size,
                              hipStream_t stream) {
  (void)in_sizes; (void)n_in; (void)out_size; (void)ws_size;
  const float* dense   = (const float*)d_in[0];
  const int*   indices = (const int*)d_in[1];
  const float* tables  = (const float*)d_in[2];
  const float* dW0 = (const float*)d_in[3];
  const float* db0 = (const float*)d_in[4];
  const float* dW1 = (const float*)d_in[5];
  const float* db1 = (const float*)d_in[6];
  const float* dW2 = (const float*)d_in[7];
  const float* db2 = (const float*)d_in[8];
  const float* oW0 = (const float*)d_in[9];
  const float* ob0 = (const float*)d_in[10];
  const float* oW1 = (const float*)d_in[11];
  const float* ob1 = (const float*)d_in[12];
  const float* oW2 = (const float*)d_in[13];
  const float* ob2 = (const float*)d_in[14];
  const float* oW3 = (const float*)d_in[15];
  const float* ob3 = (const float*)d_in[16];
  const float* oW4 = (const float*)d_in[17];
  const float* ob4 = (const float*)d_in[18];
  float* out = (float*)d_out;

  size_t off = 0;
  auto alloc16 = [&](size_t elems) -> _Float16* {
    _Float16* p = (_Float16*)((char*)d_ws + off);
    off = (off + elems * sizeof(_Float16) + 255) & ~(size_t)255;
    return p;
  };
  _Float16* x0    = alloc16((size_t)B_ * 32);
  _Float16* dW0t  = alloc16((size_t)512 * 32);
  _Float16* dW1t  = alloc16((size_t)256 * 512);
  _Float16* dW2t  = alloc16((size_t)128 * 256);
  _Float16* oW0t  = alloc16((size_t)1024 * 480);
  _Float16* oW1t  = alloc16((size_t)1024 * 1024);
  _Float16* oW2t  = alloc16((size_t)512 * 1024);
  _Float16* oW3t  = alloc16((size_t)256 * 512);
  _Float16* a1    = alloc16((size_t)B_ * 512);
  _Float16* a2    = alloc16((size_t)B_ * 256);
  _Float16* a3    = alloc16((size_t)B_ * 128);
  _Float16* z     = alloc16((size_t)B_ * 27 * 128);
  _Float16* feats = alloc16((size_t)B_ * 480);
  _Float16* o1    = alloc16((size_t)B_ * 1024);
  _Float16* o2    = alloc16((size_t)B_ * 1024);
  _Float16* o3    = alloc16((size_t)B_ * 512);
  _Float16* o4    = alloc16((size_t)B_ * 256);

  // one-time converts (tiny; live in L2 afterwards)
  convert_input_kernel<<<(B_ * 32) / 256, 256, 0, stream>>>(dense, x0);
  convert_wt_kernel<<<(512 * 32) / 256, 256, 0, stream>>>(dW0, dW0t, 13, 32, 512);
  convert_wt_kernel<<<(256 * 512) / 256, 256, 0, stream>>>(dW1, dW1t, 512, 512, 256);
  convert_wt_kernel<<<(128 * 256) / 256, 256, 0, stream>>>(dW2, dW2t, 256, 256, 128);
  convert_wt_kernel<<<(1024 * 480) / 256, 256, 0, stream>>>(oW0, oW0t, 479, 480, 1024);
  convert_wt_kernel<<<(1024 * 1024) / 256, 256, 0, stream>>>(oW1, oW1t, 1024, 1024, 1024);
  convert_wt_kernel<<<(512 * 1024) / 256, 256, 0, stream>>>(oW2, oW2t, 1024, 1024, 512);
  convert_wt_kernel<<<(256 * 512) / 256, 256, 0, stream>>>(oW3, oW3t, 512, 512, 256);

  // embedding gather (HBM-bound, ~545MB) runs before the small dense MLP GEMMs (same stream)
  embed_kernel<<<dim3(F_, B_), D_, 0, stream>>>(tables, indices, z);

  // dense MLP: 13(pad32)->512->256->128
  gemm_kernel<<<dim3(512 / 128, B_ / 128), 256, 0, stream>>>(x0, dW0t, db0, a1, 512, 32, 1);
  gemm_kernel<<<dim3(256 / 128, B_ / 128), 256, 0, stream>>>(a1, dW1t, db1, a2, 256, 512, 1);
  gemm_kernel<<<dim3(128 / 128, B_ / 128), 256, 0, stream>>>(a2, dW2t, db2, a3, 128, 256, 1);

  copy_dense_to_z<<<B_, 128, 0, stream>>>(a3, z);
  interact_kernel<<<B_ / 8, 256, 0, stream>>>(z, a3, feats);

  // over MLP: 479(pad480)->1024->1024->512->256->1
  gemm_kernel<<<dim3(1024 / 128, B_ / 128), 256, 0, stream>>>(feats, oW0t, ob0, o1, 1024, 480, 1);
  gemm_kernel<<<dim3(1024 / 128, B_ / 128), 256, 0, stream>>>(o1, oW1t, ob1, o2, 1024, 1024, 1);
  gemm_kernel<<<dim3(512 / 128, B_ / 128), 256, 0, stream>>>(o2, oW2t, ob2, o3, 512, 1024, 1);
  gemm_kernel<<<dim3(256 / 128, B_ / 128), 256, 0, stream>>>(o3, oW3t, ob3, o4, 256, 512, 1);
  final_kernel<<<B_ / 8, 256, 0, stream>>>(o4, oW4, ob4, out);
}
